// get_model_16681652978031
// MI455X (gfx1250) — matrix operand order repaired and checked by reading, not compiled, not run on hardware
//
#include <hip/hip_runtime.h>

typedef __attribute__((ext_vector_type(16))) _Float16 v16h;
typedef __attribute__((ext_vector_type(8)))  _Float16 v8h;
typedef __attribute__((ext_vector_type(8)))  float    v8f;

#define NB 8
#define NPTS 4096

// ---------------------------------------------------------------------------
// WMMA GEMM: Y[b,o,n] = res + act( bn( sum_c A[b?,o,c]*X[b,c,n] + bias ) )
// A is f16, rows padded to Cp (mult of 32), O padded to mult of 16, zero-filled
// -> A fragment = two aligned b128 loads, no masking, EXEC stays all-ones.
// One wave computes a 16(M) x 64(N) tile: 4 WMMAs per K-step reusing A.
// ---------------------------------------------------------------------------
__global__ __launch_bounds__(32)
void gemm_wmma(const _Float16* __restrict__ A, long long aBS, int Cp,
               const float* __restrict__ X, long long xBS,
               float* __restrict__ Y, long long yBS,
               _Float16* __restrict__ Yh, long long yhBS,
               const float* __restrict__ Res, long long rBS,
               const float* __restrict__ bias,
               const float* __restrict__ gamma,
               const float* __restrict__ beta,
               int O, int C, int Npix, int act)
{
    const int nBase = blockIdx.x * 64;
    const int oBase = blockIdx.y * 16;
    const int b     = blockIdx.z;
    const int lane  = threadIdx.x;
    const int lo16  = lane & 15;
    const int hiA   = (lane & 16) ? 8  : 0;   // A lane K-offset (ISA layout)
    const int hiB   = (lane & 16) ? 8  : 0;   // B lane K-offset (ISA layout)

    const _Float16* Arow = A + (size_t)b * aBS + (size_t)(oBase + lo16) * Cp + hiA;
    const float*    Bp0  = X + (size_t)b * xBS + nBase + lo16;

    v8f acc[4];
#pragma unroll
    for (int t = 0; t < 4; ++t) acc[t] = (v8f){0.f,0.f,0.f,0.f,0.f,0.f,0.f,0.f};

    const int kFull = C & ~31;
    for (int kBase = 0; kBase < kFull; kBase += 32) {
        v8h alo = *(const v8h*)(Arow + kBase);
        v8h ahi = *(const v8h*)(Arow + kBase + 16);
        v16h af = __builtin_shufflevector(alo, ahi, 0,1,2,3,4,5,6,7,8,9,10,11,12,13,14,15);
#pragma unroll
        for (int t = 0; t < 4; ++t) {
            const float* Bp = Bp0 + t * 16;
            v16h bf;
#pragma unroll
            for (int v = 0; v < 8; ++v) {
                int kb = kBase + hiB + 2 * v + ((v & 4) ? 8 : 0);
                bf[2*v]   = (_Float16)Bp[(size_t)kb     * Npix];
                bf[2*v+1] = (_Float16)Bp[(size_t)(kb+1) * Npix];
            }
            acc[t] = __builtin_amdgcn_wmma_f32_16x16x32_f16(false, af, false, bf,
                                                            (short)0, acc[t], false, false);
        }
    }
    if (C & 31) {                        // branch-free K tail (C=6, C=67 cases)
        const int kBase = kFull;
        v8h alo = *(const v8h*)(Arow + kBase);
        v8h ahi = *(const v8h*)(Arow + kBase + 16);
        v16h af = __builtin_shufflevector(alo, ahi, 0,1,2,3,4,5,6,7,8,9,10,11,12,13,14,15);
#pragma unroll
        for (int t = 0; t < 4; ++t) {
            const float* Bp = Bp0 + t * 16;
            v16h bf;
#pragma unroll
            for (int v = 0; v < 8; ++v) {
                int kb  = kBase + hiB + 2 * v + ((v & 4) ? 8 : 0);
                int k0  = (kb     < C - 1) ? kb     : C - 1;   // clamp -> v_min
                int k1  = (kb + 1 < C - 1) ? kb + 1 : C - 1;
                float b0 = Bp[(size_t)k0 * Npix];
                float b1 = Bp[(size_t)k1 * Npix];
                bf[2*v]   = (_Float16)((kb     < C) ? b0 : 0.f);  // v_cndmask
                bf[2*v+1] = (_Float16)((kb + 1 < C) ? b1 : 0.f);
            }
            acc[t] = __builtin_amdgcn_wmma_f32_16x16x32_f16(false, af, false, bf,
                                                            (short)0, acc[t], false, false);
        }
    }

    const int rOff = (lane & 16) ? 8 : 0;
    float*     Yb  = Y + (size_t)b * yBS;
    _Float16*  Yhb = Yh  ? Yh  + (size_t)b * yhBS : nullptr;
    const float* Rb = Res ? Res + (size_t)b * rBS : nullptr;
#pragma unroll
    for (int r = 0; r < 8; ++r) {
        int mo = oBase + rOff + r;
        if (mo >= O) continue;
        float bi = bias  ? bias[mo]  : 0.f;
        float ga = gamma ? gamma[mo] : 1.f;
        float be = gamma ? beta[mo]  : 0.f;
        size_t rowo = (size_t)mo * Npix + nBase + lo16;
#pragma unroll
        for (int t = 0; t < 4; ++t) {
            float v = acc[t][r] + bi;
            v = v * ga + be;
            if (act == 1)      v = fmaxf(v, 0.f);
            else if (act == 2) v = (v >= 0.f) ? v : 0.2f * v;
            if (Rb) v += Rb[rowo + t * 16];
            Yb[rowo + t * 16] = v;
            if (Yhb) Yhb[rowo + t * 16] = (_Float16)v;
        }
    }
}

// Convert f32 weight (O x C) to zero-padded f16 (Op x Cp)
__global__ void convert_weight(const float* __restrict__ W, int O, int C,
                               _Float16* __restrict__ Wh, int Op, int Cp) {
    int tid = blockIdx.x * 256 + threadIdx.x;
    if (tid >= Op * Cp) return;
    int o = tid / Cp, c = tid % Cp;
    float v = (o < O && c < C) ? W[o * C + c] : 0.f;
    Wh[tid] = (_Float16)v;
}

// ---------------------------------------------------------------------------
// Point-cloud helper kernels
// ---------------------------------------------------------------------------
__global__ void make_coords(const float* __restrict__ xyz, float* __restrict__ coords) {
    int tid = blockIdx.x * 256 + threadIdx.x;
    if (tid >= NB * NPTS) return;
    int b = tid >> 12, n = tid & (NPTS - 1);
    const float* base = xyz + (size_t)b * 6 * NPTS;
    float* o = coords + (size_t)tid * 3;
    o[0] = base[3*NPTS + n];
    o[1] = base[4*NPTS + n];
    o[2] = base[5*NPTS + n];
}

__global__ void fps_kernel(const float* __restrict__ coords, int Npts, int npoint,
                           int* __restrict__ out) {
    int b = blockIdx.x;
    const float* C = coords + (size_t)b * Npts * 3;
    __shared__ float dist[4096];
    __shared__ float redv[256];
    __shared__ int   redi[256];
    __shared__ int   curIdx;
    int t = threadIdx.x;
    for (int i = t; i < Npts; i += 256) dist[i] = 1e10f;
    if (t == 0) curIdx = 0;
    __syncthreads();
    for (int it = 0; it < npoint; ++it) {
        int far = curIdx;
        if (t == 0) out[b * npoint + it] = far;
        float cx = C[far*3], cy = C[far*3+1], cz = C[far*3+2];
        float best = -1.f; int bi = 0;
        for (int i = t; i < Npts; i += 256) {
            float dx = C[i*3]-cx, dy = C[i*3+1]-cy, dz = C[i*3+2]-cz;
            float d = dx*dx + dy*dy + dz*dz;
            float mn = fminf(dist[i], d);
            dist[i] = mn;
            if (mn > best) { best = mn; bi = i; }
        }
        redv[t] = best; redi[t] = bi;
        __syncthreads();
        for (int s = 128; s > 0; s >>= 1) {
            if (t < s && redv[t+s] > redv[t]) { redv[t] = redv[t+s]; redi[t] = redi[t+s]; }
            __syncthreads();
        }
        if (t == 0) curIdx = redi[0];
        __syncthreads();
    }
}

__global__ void gather_coords(const float* __restrict__ in, const int* __restrict__ idx,
                              int Nin, int Nout, float* __restrict__ out) {
    int tid = blockIdx.x * 256 + threadIdx.x;
    if (tid >= NB * Nout) return;
    int b = tid / Nout;
    int s = idx[tid];
    const float* p = in + ((size_t)b * Nin + s) * 3;
    float* o = out + (size_t)tid * 3;
    o[0] = p[0]; o[1] = p[1]; o[2] = p[2];
}

__global__ void ball_query_kernel(const float* __restrict__ coords,
                                  const float* __restrict__ newc,
                                  int Npts, int np, float r2,
                                  int* __restrict__ gidx) {
    int tid = blockIdx.x * 256 + threadIdx.x;
    if (tid >= NB * np) return;
    int b = tid / np;
    const float* C = coords + (size_t)b * Npts * 3;
    const float* q = newc + (size_t)tid * 3;
    float cx = q[0], cy = q[1], cz = q[2];
    int* out = gidx + (size_t)tid * 32;
    int cnt = 0, first = -1;
    for (int i = 0; i < Npts && cnt < 32; ++i) {
        float dx = C[i*3]-cx, dy = C[i*3+1]-cy, dz = C[i*3+2]-cz;
        if (dx*dx + dy*dy + dz*dz <= r2) {
            if (first < 0) first = i;
            out[cnt++] = i;
        }
    }
    if (first < 0) first = Npts - 1;
    for (; cnt < 32; ++cnt) out[cnt] = first;
}

__global__ void group_kernel(const float* __restrict__ coords,
                             const float* __restrict__ newc,
                             const float* __restrict__ pts, long long ptsBS,
                             int Cpts, int Npts, int np,
                             const int* __restrict__ gidx,
                             float* __restrict__ gout) {
    int tid = blockIdx.x * 256 + threadIdx.x;
    int tot = NB * np * 32;
    if (tid >= tot) return;
    int b = tid / (np * 32);
    int rem = tid % (np * 32);
    int j = rem / 32, k = rem % 32;
    int src = gidx[((size_t)b * np + j) * 32 + k];
    size_t pix  = (size_t)k * np + j;
    size_t pixs = (size_t)32 * np;
    float* base = gout + (size_t)b * (3 + Cpts) * pixs;
    const float* cw = coords + ((size_t)b * Npts + src) * 3;
    const float* cc = newc   + ((size_t)b * np   + j  ) * 3;
    base[0*pixs + pix] = cw[0] - cc[0];
    base[1*pixs + pix] = cw[1] - cc[1];
    base[2*pixs + pix] = cw[2] - cc[2];
    const float* pb = pts + (size_t)b * ptsBS;
    for (int c = 0; c < Cpts; ++c)
        base[(size_t)(3 + c) * pixs + pix] = pb[(size_t)c * Npts + src];
}

__global__ void maxpool_k(const float* __restrict__ in, long long iBS,
                          int O, int np, float* __restrict__ out) {
    int tid = blockIdx.x * 256 + threadIdx.x;
    if (tid >= NB * O * np) return;
    int b = tid / (O * np);
    int rem = tid % (O * np);
    int o = rem / np, j = rem % np;
    const float* p = in + (size_t)b * iBS + (size_t)o * 32 * np;
    float m = -3.4e38f;
    for (int k = 0; k < 32; ++k) m = fmaxf(m, p[(size_t)k * np + j]);
    out[((size_t)b * O + o) * np + j] = m;
}

// ---------------------------------------------------------------------------
// Attention helpers
// ---------------------------------------------------------------------------
__global__ void attn_energy_softmax(const float* __restrict__ xk, float* __restrict__ att) {
    int b = blockIdx.x >> 8;
    int n = blockIdx.x & 255;
    int m = threadIdx.x;
    const float* K = xk + (size_t)b * 32 * 256;
    float e = 0.f;
#pragma unroll
    for (int o = 0; o < 32; ++o) e += K[o*256 + n] * K[o*256 + m];
    __shared__ float red[256];
    red[m] = e; __syncthreads();
    for (int s = 128; s > 0; s >>= 1) { if (m < s) red[m] = fmaxf(red[m], red[m+s]); __syncthreads(); }
    float mx = red[0]; __syncthreads();
    float ex = expf(e - mx);
    red[m] = ex; __syncthreads();
    for (int s = 128; s > 0; s >>= 1) { if (m < s) red[m] += red[m+s]; __syncthreads(); }
    att[(((size_t)b * 256) + n) * 256 + m] = ex / red[0];
}

__global__ void attn_colnorm(float* __restrict__ att) {
    int tid = blockIdx.x * 256 + threadIdx.x;
    if (tid >= NB * 256) return;
    int b = tid / 256, m = tid % 256;
    float* base = att + (size_t)b * 256 * 256 + m;
    float s = 0.f;
    for (int n = 0; n < 256; ++n) s += base[(size_t)n * 256];
    float inv = 1.f / (1e-9f + s);
    for (int n = 0; n < 256; ++n) base[(size_t)n * 256] *= inv;
}

__global__ void sub_k(const float* __restrict__ x, long long xBS, float* __restrict__ y) {
    int tid = blockIdx.x * 256 + threadIdx.x;
    if (tid >= NB * 128 * 256) return;
    int b = tid / (128 * 256);
    int rem = tid % (128 * 256);
    y[tid] = x[(size_t)b * xBS + rem] - y[tid];
}

// ---------------------------------------------------------------------------
// Feature propagation helpers
// ---------------------------------------------------------------------------
__global__ void rowmax_k(const float* __restrict__ in, long long iBS, int Cs, int Npix,
                         float* __restrict__ out) {
    int tid = blockIdx.x * 256 + threadIdx.x;
    if (tid >= NB * Cs) return;
    int b = tid / Cs, c = tid % Cs;
    const float* p = in + (size_t)b * iBS + (size_t)c * Npix;
    float m = -3.4e38f;
    for (int n = 0; n < Npix; ++n) m = fmaxf(m, p[n]);
    out[tid] = m;
}

__global__ void copy_rows(const float* __restrict__ in, long long iBS,
                          float* __restrict__ out, long long oBS,
                          int rowOff, int Cs, int Npix) {
    int tid = blockIdx.x * 256 + threadIdx.x;
    if (tid >= NB * Cs * Npix) return;
    int b = tid / (Cs * Npix);
    int rem = tid % (Cs * Npix);
    int c = rem / Npix, n = rem % Npix;
    out[(size_t)b * oBS + (size_t)(rowOff + c) * Npix + n] =
        in[(size_t)b * iBS + (size_t)c * Npix + n];
}

__global__ void bcast_rows(const float* __restrict__ vals,
                           float* __restrict__ out, long long oBS,
                           int rowOff, int Cs, int Npix) {
    int tid = blockIdx.x * 256 + threadIdx.x;
    if (tid >= NB * Cs * Npix) return;
    int b = tid / (Cs * Npix);
    int rem = tid % (Cs * Npix);
    int c = rem / Npix, n = rem % Npix;
    out[(size_t)b * oBS + (size_t)(rowOff + c) * Npix + n] = vals[b * Cs + c];
}

__global__ void knn3_kernel(const float* __restrict__ tgt, const float* __restrict__ src,
                            int Nt, int Ns, int* __restrict__ idx, float* __restrict__ w) {
    int tid = blockIdx.x * 256 + threadIdx.x;
    if (tid >= NB * Nt) return;
    int b = tid / Nt;
    const float* T = tgt + (size_t)tid * 3;
    const float* S = src + (size_t)b * Ns * 3;
    float tx = T[0], ty = T[1], tz = T[2];
    float d0 = 3.4e38f, d1 = 3.4e38f, d2 = 3.4e38f;
    int i0 = 0, i1 = 0, i2 = 0;
    for (int i = 0; i < Ns; ++i) {
        float dx = tx - S[i*3], dy = ty - S[i*3+1], dz = tz - S[i*3+2];
        float d = dx*dx + dy*dy + dz*dz;
        if (d < d0)      { d2=d1; i2=i1; d1=d0; i1=i0; d0=d; i0=i; }
        else if (d < d1) { d2=d1; i2=i1; d1=d; i1=i; }
        else if (d < d2) { d2=d; i2=i; }
    }
    float w0 = 1.f/(d0+1e-8f), w1 = 1.f/(d1+1e-8f), w2 = 1.f/(d2+1e-8f);
    float s = w0 + w1 + w2;
    size_t o = (size_t)tid * 3;
    idx[o] = i0; idx[o+1] = i1; idx[o+2] = i2;
    w[o] = w0/s; w[o+1] = w1/s; w[o+2] = w2/s;
}

__global__ void interp_apply(const float* __restrict__ src, long long sBS, int Cs, int Ns,
                             const int* __restrict__ idx, const float* __restrict__ w, int Nt,
                             float* __restrict__ out, long long oBS, int rowOff) {
    int tid = blockIdx.x * 256 + threadIdx.x;
    if (tid >= NB * Cs * Nt) return;
    int b = tid / (Cs * Nt);
    int rem = tid % (Cs * Nt);
    int c = rem / Nt, n = rem % Nt;
    size_t ko = ((size_t)b * Nt + n) * 3;
    const float* S = src + (size_t)b * sBS + (size_t)c * Ns;
    out[(size_t)b * oBS + (size_t)(rowOff + c) * Nt + n] =
        w[ko] * S[idx[ko]] + w[ko+1] * S[idx[ko+1]] + w[ko+2] * S[idx[ko+2]];
}

__global__ void logsoftmax_out(const float* __restrict__ logits, float* __restrict__ out) {
    int tid = blockIdx.x * 256 + threadIdx.x;
    if (tid >= NB * NPTS) return;
    int b = tid >> 12, n = tid & (NPTS - 1);
    const float* p = logits + (size_t)b * 13 * NPTS + n;
    float v[13], mx = -3.4e38f;
#pragma unroll
    for (int k = 0; k < 13; ++k) { v[k] = p[(size_t)k * NPTS]; mx = fmaxf(mx, v[k]); }
    float s = 0.f;
#pragma unroll
    for (int k = 0; k < 13; ++k) s += expf(v[k] - mx);
    float lse = mx + logf(s);
    float* o = out + (size_t)tid * 13;
#pragma unroll
    for (int k = 0; k < 13; ++k) o[k] = v[k] - lse;
}

// ---------------------------------------------------------------------------
extern "C" void kernel_launch(void* const* d_in, const int* in_sizes, int n_in,
                              void* d_out, int out_size, void* d_ws, size_t ws_size,
                              hipStream_t stream) {
    (void)in_sizes; (void)n_in; (void)out_size; (void)ws_size;
#define P(i) ((const float*)d_in[(i)])
    const float* xyz = P(87);

    char* wsp = (char*)d_ws;
    auto alloc = [&](size_t bytes) -> void* {
        void* p = (void*)wsp;
        wsp += (bytes + 255) & ~(size_t)255;
        return p;
    };
    float* coords0   = (float*)alloc((size_t)NB*NPTS*3*4);
    int*   fidx1     = (int*)  alloc((size_t)NB*1024*4);
    float* coords1   = (float*)alloc((size_t)NB*1024*3*4);
    int*   gidx1     = (int*)  alloc((size_t)NB*1024*32*4);
    float* g_in1     = (float*)alloc((size_t)NB*6*32768*4);
    float* bufA      = (float*)alloc((size_t)NB*64*32768*4);
    float* bufB      = (float*)alloc((size_t)NB*64*32768*4);
    float* l1_points = (float*)alloc((size_t)NB*64*1024*4);
    int*   fidx2     = (int*)  alloc((size_t)NB*256*4);
    float* coords2   = (float*)alloc((size_t)NB*256*3*4);
    int*   gidx2     = (int*)  alloc((size_t)NB*256*32*4);
    float* g_in2     = (float*)alloc((size_t)NB*67*8192*4);
    float* l2_points = (float*)alloc((size_t)NB*128*256*4);
    float* attcat    = (float*)alloc((size_t)NB*640*256*4);
    float* xbuf      = (float*)alloc((size_t)NB*128*256*4);
    float* xk        = (float*)alloc((size_t)NB*32*256*4);
    float* xv        = (float*)alloc((size_t)NB*128*256*4);
    _Float16* xvh    = (_Float16*)alloc((size_t)NB*128*256*2);
    float* attbuf    = (float*)alloc((size_t)NB*256*256*4);
    float* xtmp      = (float*)alloc((size_t)NB*128*256*4);
    float* gmaxb     = (float*)alloc((size_t)NB*512*4);
    float* fp2_in    = (float*)alloc((size_t)NB*1088*1024*4);
    int*   ki2       = (int*)  alloc((size_t)NB*1024*3*4);
    float* kw2       = (float*)alloc((size_t)NB*1024*3*4);
    float* l1p       = (float*)alloc((size_t)NB*128*1024*4);
    int*   ki1       = (int*)  alloc((size_t)NB*NPTS*3*4);
    float* kw1       = (float*)alloc((size_t)NB*NPTS*3*4);
    float* fp1_in    = (float*)alloc((size_t)NB*128*NPTS*4);
    float* logits    = (float*)alloc((size_t)NB*13*NPTS*4);

    float* out_seg   = (float*)d_out;                    // (B,N,13) log-probs
    float* l2p       = out_seg + (size_t)NB*NPTS*13;     // (B,512,256)

    auto g1 = [](int total) { return dim3((total + 255) / 256); };

    // pre-convert all static weights to padded f16
    auto convW = [&](int idx, int O, int C) -> const _Float16* {
        int Op = (O + 15) & ~15;
        int Cp = (C + 31) & ~31;
        _Float16* p = (_Float16*)alloc((size_t)Op * Cp * 2);
        convert_weight<<<g1(Op*Cp), 256, 0, stream>>>(P(idx), O, C, p, Op, Cp);
        return p;
    };
    const _Float16* w_sa1[3] = { convW(63,32,6),   convW(67,32,32),  convW(71,64,32)  };
    const _Float16* w_sa2[3] = { convW(75,64,67),  convW(79,64,64),  convW(83,128,64) };
    const _Float16* w_att1 = convW(0,128,128);
    const _Float16* w_att2 = convW(1,128,128);
    const _Float16* w_q[4]; const _Float16* w_t[4]; const _Float16* w_v[4];
    for (int i = 0; i < 4; ++i) {
        int sb = 6 + 7*i;
        w_q[i] = convW(sb+0, 32, 128);
        w_t[i] = convW(sb+1, 128, 128);
        w_v[i] = convW(sb+2, 128, 128);
    }
    const _Float16* w_fuse = convW(54, 512, 640);
    const _Float16* w_fp2a = convW(46, 256, 1088);
    const _Float16* w_fp2b = convW(50, 128, 256);
    const _Float16* w_fp1[3] = { convW(34,128,128), convW(38,128,128), convW(42,128,128) };
    const _Float16* w_h1 = convW(57, 128, 128);
    const _Float16* w_h2 = convW(58, 13, 128);

    auto gemm = [&](const _Float16* A, long long aBS, const float* X, long long xBS,
                    float* Y, long long yBS, _Float16* Yh, long long yhBS,
                    const float* Res, long long rBS,
                    const float* bias, const float* ga, const float* be,
                    int O, int C, int Npix, int act) {
        int Cp = (C + 31) & ~31;
        dim3 gr(Npix / 64, (O + 15) / 16, NB);
        gemm_wmma<<<gr, dim3(32), 0, stream>>>(A, aBS, Cp, X, xBS, Y, yBS, Yh, yhBS,
                                               Res, rBS, bias, ga, be, O, C, Npix, act);
    };

    // ---- Set abstraction 1 (4096 -> 1024, r=0.1, 32 samples) ----
    make_coords<<<g1(NB*NPTS), 256, 0, stream>>>(xyz, coords0);
    fps_kernel<<<NB, 256, 0, stream>>>(coords0, NPTS, 1024, fidx1);
    gather_coords<<<g1(NB*1024), 256, 0, stream>>>(coords0, fidx1, NPTS, 1024, coords1);
    ball_query_kernel<<<g1(NB*1024), 256, 0, stream>>>(coords0, coords1, NPTS, 1024, 0.01f, gidx1);
    group_kernel<<<g1(NB*1024*32), 256, 0, stream>>>(coords0, coords1,
        xyz + 3*NPTS, (long long)6*NPTS, 3, NPTS, 1024, gidx1, g_in1);
    gemm(w_sa1[0], 0, g_in1, 6LL*32768, bufA, 32LL*32768, nullptr,0, nullptr,0, P(64), P(66), P(65), 32, 6, 32768, 1);
    gemm(w_sa1[1], 0, bufA, 32LL*32768, bufB, 32LL*32768, nullptr,0, nullptr,0, P(68), P(70), P(69), 32, 32, 32768, 1);
    gemm(w_sa1[2], 0, bufB, 32LL*32768, bufA, 64LL*32768, nullptr,0, nullptr,0, P(72), P(74), P(73), 64, 32, 32768, 1);
    maxpool_k<<<g1(NB*64*1024), 256, 0, stream>>>(bufA, 64LL*32768, 64, 1024, l1_points);

    // ---- Set abstraction 2 (1024 -> 256, r=0.2, 32 samples) ----
    fps_kernel<<<NB, 256, 0, stream>>>(coords1, 1024, 256, fidx2);
    gather_coords<<<g1(NB*256), 256, 0, stream>>>(coords1, fidx2, 1024, 256, coords2);
    ball_query_kernel<<<g1(NB*256), 256, 0, stream>>>(coords1, coords2, 1024, 256, 0.04f, gidx2);
    group_kernel<<<g1(NB*256*32), 256, 0, stream>>>(coords1, coords2,
        l1_points, 64LL*1024, 64, 1024, 256, gidx2, g_in2);
    gemm(w_sa2[0], 0, g_in2, 67LL*8192, bufA, 64LL*8192, nullptr,0, nullptr,0, P(76), P(78), P(77), 64, 67, 8192, 1);
    gemm(w_sa2[1], 0, bufA, 64LL*8192, bufB, 64LL*8192, nullptr,0, nullptr,0, P(80), P(82), P(81), 64, 64, 8192, 1);
    gemm(w_sa2[2], 0, bufB, 64LL*8192, bufA, 128LL*8192, nullptr,0, nullptr,0, P(84), P(86), P(85), 128, 64, 8192, 1);
    maxpool_k<<<g1(NB*128*256), 256, 0, stream>>>(bufA, 128LL*8192, 128, 256, l2_points);

    // ---- Stacked attention (B,128,256) ----
    gemm(w_att1, 0, l2_points, 128LL*256, xtmp, 128LL*256, nullptr,0, nullptr,0, nullptr, P(4), P(2), 128, 128, 256, 1);
    gemm(w_att2, 0, xtmp, 128LL*256, xbuf, 128LL*256, nullptr,0, nullptr,0, nullptr, P(5), P(3), 128, 128, 256, 1);
    const float* xcur = xbuf;
    long long xcurBS = 128LL * 256;
    for (int i = 0; i < 4; ++i) {
        int sb = 6 + 7 * i;   // Wq,Wt,Wv,bt,btr,bv,g
        gemm(w_q[i], 0, xcur, xcurBS, xk, 32LL*256, nullptr,0, nullptr,0, nullptr, nullptr, nullptr, 32, 128, 256, 0);
        attn_energy_softmax<<<NB*256, 256, 0, stream>>>(xk, attbuf);
        attn_colnorm<<<g1(NB*256), 256, 0, stream>>>(attbuf);
        gemm(w_v[i], 0, xcur, xcurBS, xv, 128LL*256, xvh, 128LL*256, nullptr,0, P(sb+5), nullptr, nullptr, 128, 128, 256, 0);
        // xr = xv @ att (A = batched f16 copy emitted by previous gemm)
        gemm(xvh, 128LL*256, attbuf, 256LL*256, xtmp, 128LL*256, nullptr,0, nullptr,0, nullptr, nullptr, nullptr, 128, 256, 256, 0);
        sub_k<<<g1(NB*128*256), 256, 0, stream>>>(xcur, xcurBS, xtmp);  // xtmp = x - xr
        float* xnext = attcat + (size_t)i * 128 * 256;
        gemm(w_t[i], 0, xtmp, 128LL*256, xnext, 640LL*256, nullptr,0, xcur, xcurBS,
             P(sb+4), P(sb+6), P(sb+3), 128, 128, 256, 1);
        xcur = xnext; xcurBS = 640LL * 256;
    }
    copy_rows<<<g1(NB*128*256), 256, 0, stream>>>(l2_points, 128LL*256, attcat, 640LL*256, 512, 128, 256);

    // ---- Fuse: 640 -> 512, leaky_relu(0.2); writes l2p straight into d_out ----
    gemm(w_fuse, 0, attcat, 640LL*256, l2p, 512LL*256, nullptr,0, nullptr,0, nullptr, P(56), P(55), 512, 640, 256, 2);

    // ---- fp2: interpolate (256 -> 1024 points), channels 64+512+512 = 1088 ----
    rowmax_k<<<g1(NB*512), 256, 0, stream>>>(l2p, 512LL*256, 512, 256, gmaxb);
    copy_rows<<<g1(NB*64*1024), 256, 0, stream>>>(l1_points, 64LL*1024, fp2_in, 1088LL*1024, 0, 64, 1024);
    bcast_rows<<<g1(NB*512*1024), 256, 0, stream>>>(gmaxb, fp2_in, 1088LL*1024, 64, 512, 1024);
    knn3_kernel<<<g1(NB*1024), 256, 0, stream>>>(coords1, coords2, 1024, 256, ki2, kw2);
    interp_apply<<<g1(NB*512*1024), 256, 0, stream>>>(l2p, 512LL*256, 512, 256, ki2, kw2, 1024,
                                                      fp2_in, 1088LL*1024, 576);
    gemm(w_fp2a, 0, fp2_in, 1088LL*1024, bufA, 256LL*1024, nullptr,0, nullptr,0, P(47), P(49), P(48), 256, 1088, 1024, 1);
    gemm(w_fp2b, 0, bufA, 256LL*1024, l1p, 128LL*1024, nullptr,0, nullptr,0, P(51), P(53), P(52), 128, 256, 1024, 1);

    // ---- fp1: interpolate (1024 -> 4096 points), 3x 128->128 ----
    knn3_kernel<<<g1(NB*NPTS), 256, 0, stream>>>(coords0, coords1, NPTS, 1024, ki1, kw1);
    interp_apply<<<g1(NB*128*NPTS), 256, 0, stream>>>(l1p, 128LL*1024, 128, 1024, ki1, kw1, NPTS,
                                                      fp1_in, 128LL*NPTS, 0);
    gemm(w_fp1[0], 0, fp1_in, 128LL*NPTS, bufA, 128LL*NPTS, nullptr,0, nullptr,0, P(35), P(37), P(36), 128, 128, NPTS, 1);
    gemm(w_fp1[1], 0, bufA, 128LL*NPTS, bufB, 128LL*NPTS, nullptr,0, nullptr,0, P(39), P(41), P(40), 128, 128, NPTS, 1);
    gemm(w_fp1[2], 0, bufB, 128LL*NPTS, bufA, 128LL*NPTS, nullptr,0, nullptr,0, P(43), P(45), P(44), 128, 128, NPTS, 1);

    // ---- Head: 128->128 relu, 128->13, log_softmax + transpose ----
    gemm(w_h1, 0, bufA, 128LL*NPTS, bufB, 128LL*NPTS, nullptr,0, nullptr,0, P(59), P(62), P(61), 128, 128, NPTS, 1);
    gemm(w_h2, 0, bufB, 128LL*NPTS, logits, 13LL*NPTS, nullptr,0, nullptr,0, P(60), nullptr, nullptr, 13, 128, NPTS, 0);
    logsoftmax_out<<<g1(NB*NPTS), 256, 0, stream>>>(logits, out_seg);
#undef P
}
